// CustomAttentionLayer_28441273434636
// MI455X (gfx1250) — matrix operand order repaired and checked by reading, not compiled, not run on hardware
//
#include <hip/hip_runtime.h>
#include <hip/hip_bf16.h>

typedef _Float16 f16;
typedef __attribute__((ext_vector_type(16))) _Float16 v16h;
typedef __attribute__((ext_vector_type(8)))  _Float16 v8h;
typedef __attribute__((ext_vector_type(8)))  float    v8f;
typedef __attribute__((ext_vector_type(4)))  float    v4f;

#define S_LEN 4096
#define D_DIM 512
#define P_DIM 512
#define BATCH 4
#define ROWS_TOTAL (BATCH * S_LEN)   // 16384
#define QKV_COLS (3 * P_DIM)         // 1536
#define KSTAGE 128                   // K-dim halves staged per async chunk

// ---------------------------------------------------------------------------
// CDNA5 async Global->LDS copy (ASYNCcnt-tracked, CDNA5 ISA)
// ---------------------------------------------------------------------------
static __device__ __forceinline__ void async_load_b128_to_lds(uint32_t ldsAddr,
                                                              const void* gaddr) {
  asm volatile("global_load_async_to_lds_b128 %0, %1, off"
               :: "v"(ldsAddr), "v"(gaddr)
               : "memory");
}
#define WAIT_ASYNCCNT_0() asm volatile("s_wait_asynccnt 0x0" ::: "memory")
#define WAIT_ASYNCCNT_4() asm volatile("s_wait_asynccnt 0x4" ::: "memory")

// ---------------------------------------------------------------------------
// Fragment loaders per CDNA5 ISA VGPR layouts (wave32).
// A (16x32 f16): lane L holds row m=L&15; h[0..7]=A[m][k0..k0+7],
//                h[8..15]=A[m][k0+16..k0+23], k0 = kChunk + (L>>4)*8.
// B (32x16 f16): lane L holds col n=L&15; h[0..7]=B[k0..k0+7][n],
//                h[8..15]=B[k0+16..k0+23][n], k0 = kChunk + (L>>4)*8.
//                With B stored transposed (Bt[n][k]) these are two 8-half runs.
// ---------------------------------------------------------------------------
static __device__ __forceinline__ v16h load_a_frag(const f16* __restrict__ rowPtr,
                                                   int kChunk, int lane) {
  const int k0 = kChunk + ((lane >> 4) << 3);
  v8h lo = *(const v8h*)(rowPtr + k0);
  v8h hi = *(const v8h*)(rowPtr + k0 + 16);
  v16h r;
#pragma unroll
  for (int i = 0; i < 8; ++i) { r[i] = lo[i]; r[i + 8] = hi[i]; }
  return r;
}

static __device__ __forceinline__ v16h load_bt_frag(const f16* rowPtr,
                                                    int kChunk, int lane) {
  const int k0 = kChunk + ((lane >> 4) << 3);
  v8h lo = *(const v8h*)(rowPtr + k0);
  v8h hi = *(const v8h*)(rowPtr + k0 + 16);
  v16h r;
#pragma unroll
  for (int i = 0; i < 8; ++i) { r[i] = lo[i]; r[i + 8] = hi[i]; }
  return r;
}

static __device__ __forceinline__ v8f wmma_f16(v16h a, v16h b, v8f c) {
  return __builtin_amdgcn_wmma_f32_16x16x32_f16(false, a, false, b,
                                                (short)0, c, false, false);
}

// ---------------------------------------------------------------------------
// Kernel 1: x fp32 -> f16 (4 elements / thread)
// ---------------------------------------------------------------------------
__global__ void prep_x(const float* __restrict__ x, f16* __restrict__ xh) {
  const int i = (blockIdx.x * blockDim.x + threadIdx.x) * 4;
  v4f v = *(const v4f*)(x + i);
  f16 h0 = (f16)v[0], h1 = (f16)v[1], h2 = (f16)v[2], h3 = (f16)v[3];
  xh[i + 0] = h0; xh[i + 1] = h1; xh[i + 2] = h2; xh[i + 3] = h3;
}

// ---------------------------------------------------------------------------
// Kernel 2: weights fp32 -> f16, transposed.
//   WqkvT[(sec*512 + p)][k] = W{q,k,v}[k][p]
//   WoT[d][p]              = Wo[p][d]
// ---------------------------------------------------------------------------
__global__ void prep_weights(const float* __restrict__ Wq, const float* __restrict__ Wk,
                             const float* __restrict__ Wv, const float* __restrict__ Wo,
                             f16* __restrict__ WqkvT, f16* __restrict__ WoT) {
  const int idx = blockIdx.x * blockDim.x + threadIdx.x;   // 0 .. 512*512-1
  const int k = idx >> 9;        // row of Wq/Wk/Wv (D dim)
  const int p = idx & 511;       // col (P dim)
  WqkvT[((size_t)(0 * P_DIM + p)) * D_DIM + k] = (f16)Wq[idx];
  WqkvT[((size_t)(1 * P_DIM + p)) * D_DIM + k] = (f16)Wk[idx];
  WqkvT[((size_t)(2 * P_DIM + p)) * D_DIM + k] = (f16)Wv[idx];
  const int pr = idx >> 9;       // row of Wo (P dim)
  const int d  = idx & 511;      // col (D dim)
  WoT[((size_t)d) * P_DIM + pr] = (f16)Wo[idx];
}

// ---------------------------------------------------------------------------
// Kernel 3: fused QKV GEMM  [16384,512] x [512,1536].
// Block = 256 threads = 8 waves; block tile = 32 rows x 256 cols.
// q,k stored row-major f16; v stored transposed (vT[b][p][t]) with
// contiguous 16-byte stores per lane.
// ---------------------------------------------------------------------------
__global__ void qkv_gemm(const f16* __restrict__ xh, const f16* __restrict__ WqkvT,
                         const float* __restrict__ bq, const float* __restrict__ bk,
                         const float* __restrict__ bv,
                         f16* __restrict__ qh, f16* __restrict__ kh, f16* __restrict__ vT) {
  const int lane = threadIdx.x & 31;
  const int wave = threadIdx.x >> 5;
  const int rowBase = blockIdx.y * 32 + (wave & 1) * 16;        // global row
  const int colBase = blockIdx.x * 256 + (wave >> 1) * 64;      // 0..1535

  v8f acc[4] = {v8f{}, v8f{}, v8f{}, v8f{}};
  const f16* aRow = xh + (size_t)(rowBase + (lane & 15)) * D_DIM;

  for (int kc = 0; kc < D_DIM; kc += 32) {
    v16h a = load_a_frag(aRow, kc, lane);
#pragma unroll
    for (int j = 0; j < 4; ++j) {
      const f16* bRow = WqkvT + (size_t)(colBase + j * 16 + (lane & 15)) * D_DIM;
      v16h b = load_bt_frag(bRow, kc, lane);
      acc[j] = wmma_f16(a, b, acc[j]);
    }
  }

  const int mBase = rowBase + ((lane >> 4) << 3);
  const int nLane = lane & 15;
#pragma unroll
  for (int j = 0; j < 4; ++j) {
    const int col = colBase + j * 16 + nLane;
    if (col < 1024) {
      const float bias = (col < 512) ? bq[col] : bk[col - 512];
      f16* dst = (col < 512) ? qh : kh;
      const int p = col & 511;
#pragma unroll
      for (int r = 0; r < 8; ++r)
        dst[(size_t)(mBase + r) * P_DIM + p] = (f16)(acc[j][r] + bias);
    } else {
      const int p = col - 1024;
      const float bias = bv[p];
      const int b = mBase >> 12;        // / 4096
      const int t = mBase & 4095;
      v8h pack;
#pragma unroll
      for (int r = 0; r < 8; ++r) pack[r] = (f16)(acc[j][r] + bias);
      *(v8h*)(vT + ((size_t)b * P_DIM + p) * S_LEN + t) = pack;
    }
  }
}

// ---------------------------------------------------------------------------
// Kernel 4: flash attention.  Block = 8 waves, 32 query rows per block.
// Score-phase K tiles are staged Global->LDS with double-buffered
// GLOBAL_LOAD_ASYNC_TO_LDS_B128 (ASYNCcnt), B-fragments then come from LDS
// via ds_load_b128; V fragments stream from L2 (vT layout, contiguous b128).
// Online softmax by threads 0..31 (one row each) through LDS.
// ---------------------------------------------------------------------------
__global__ void attention(const f16* __restrict__ qh, const f16* __restrict__ kh,
                          const f16* __restrict__ vT, f16* __restrict__ attH) {
  __shared__ __align__(16) f16   sK[2][64][KSTAGE + 8];   // ~34.8 KB
  __shared__ __align__(16) float sScores[32][66];          //  8.4 KB
  __shared__ __align__(16) f16   sAttn[32][72];            //  4.6 KB
  __shared__ float sRowScale[32];

  const int lane = threadIdx.x & 31;
  const int wave = threadIdx.x >> 5;
  const int b  = blockIdx.y;
  const int s0 = blockIdx.x * 32;
  const int r0 = (wave & 1) * 16;
  const int cq = wave >> 1;                 // 0..3
  const int hi8 = (lane >> 4) << 3;

  const f16* qRow = qh + (size_t)(b * S_LEN + s0 + r0 + (lane & 15)) * P_DIM;

  // Per-thread async staging slice: row = tid>>2 (0..63), 4x b128 each.
  const int stRow = threadIdx.x >> 2;
  const int stCol = (threadIdx.x & 3) * 32;

  v8f acc[8] = {v8f{}, v8f{}, v8f{}, v8f{}, v8f{}, v8f{}, v8f{}, v8f{}};
  float rowM = -__builtin_inff();
  float rowL = 0.0f;
  const float scale = 0.125f;               // 1/sqrt(KEY_DIM=64)

  for (int t0 = 0; t0 < S_LEN; t0 += 64) {
    if (t0 + 64 < S_LEN)
      __builtin_prefetch(kh + (size_t)(b * S_LEN + t0 + 64 + cq * 16 + (lane & 15)) * P_DIM, 0, 1);

    // ---- stage first K chunk of this key tile ----
    {
      const f16* g = kh + (size_t)(b * S_LEN + t0 + stRow) * P_DIM + stCol;
      const uint32_t l = (uint32_t)(uintptr_t)&sK[0][stRow][stCol];
#pragma unroll
      for (int c = 0; c < 4; ++c)
        async_load_b128_to_lds(l + c * 16, g + c * 8);
    }

    // ---- scores: Q (16x512) . K^T (512x16), double-buffered LDS K ----
    v8f sc = v8f{};
#pragma unroll
    for (int st = 0; st < P_DIM / KSTAGE; ++st) {   // 4 stages
      const int pc0 = st * KSTAGE;
      __syncthreads();   // all waves done reading the buffer we are about to refill
      if (st + 1 < P_DIM / KSTAGE) {
        const f16* g = kh + (size_t)(b * S_LEN + t0 + stRow) * P_DIM + pc0 + KSTAGE + stCol;
        const uint32_t l = (uint32_t)(uintptr_t)&sK[(st + 1) & 1][stRow][stCol];
#pragma unroll
        for (int c = 0; c < 4; ++c)
          async_load_b128_to_lds(l + c * 16, g + c * 8);
        WAIT_ASYNCCNT_4();   // previous chunk's 4 loads complete (in-order), next 4 in flight
      } else {
        WAIT_ASYNCCNT_0();
      }
      __syncthreads();   // current chunk visible to every wave
      const f16* kRow = &sK[st & 1][cq * 16 + (lane & 15)][0];
#pragma unroll
      for (int pcl = 0; pcl < KSTAGE; pcl += 32) {
        v16h a  = load_a_frag(qRow, pc0 + pcl, lane);
        v16h kb = load_bt_frag(kRow, pcl, lane);
        sc = wmma_f16(a, kb, sc);
      }
    }
    {
      const int n  = cq * 16 + (lane & 15);
      const int mB = r0 + hi8;
#pragma unroll
      for (int r = 0; r < 8; ++r) sScores[mB + r][n] = sc[r] * scale;
    }
    __syncthreads();

    // ---- online softmax: one row per thread (threads 0..31) ----
    if (threadIdx.x < 32) {
      const int m = threadIdx.x;
      float mt = -__builtin_inff();
      for (int t = 0; t < 64; ++t) mt = fmaxf(mt, sScores[m][t]);
      const float Mn   = fmaxf(rowM, mt);
      const float corr = __expf(rowM - Mn);
      float sum = 0.0f;
      for (int t = 0; t < 64; ++t) {
        const float e = __expf(sScores[m][t] - Mn);
        sAttn[m][t] = (f16)e;
        sum += e;
      }
      rowL = rowL * corr + sum;
      rowM = Mn;
      sRowScale[m] = corr;
    }
    __syncthreads();

    // ---- rescale accumulators, then PV: attn(16x64) . V(64x128) ----
    float cs[8];
#pragma unroll
    for (int r = 0; r < 8; ++r) cs[r] = sRowScale[r0 + hi8 + r];
#pragma unroll
    for (int j = 0; j < 8; ++j)
#pragma unroll
      for (int r = 0; r < 8; ++r) acc[j][r] *= cs[r];

#pragma unroll
    for (int tc = 0; tc < 64; tc += 32) {
      const int m  = r0 + (lane & 15);
      const int k0 = tc + hi8;
      v8h alo = *(const v8h*)&sAttn[m][k0];
      v8h ahi = *(const v8h*)&sAttn[m][k0 + 16];
      v16h a;
#pragma unroll
      for (int i = 0; i < 8; ++i) { a[i] = alo[i]; a[i + 8] = ahi[i]; }
#pragma unroll
      for (int j = 0; j < 8; ++j) {
        const f16* vRow = vT + ((size_t)b * P_DIM + cq * 128 + j * 16 + (lane & 15)) * S_LEN + t0;
        v16h vb = load_bt_frag(vRow, tc, lane);
        acc[j] = wmma_f16(a, vb, acc[j]);
      }
    }
    __syncthreads();
  }

  // ---- final 1/l normalization, store attended as f16 row-major ----
  if (threadIdx.x < 32) sRowScale[threadIdx.x] = 1.0f / rowL;
  __syncthreads();
  float cs[8];
#pragma unroll
  for (int r = 0; r < 8; ++r) cs[r] = sRowScale[r0 + hi8 + r];
  const int gRow = b * S_LEN + s0 + r0 + hi8;
#pragma unroll
  for (int j = 0; j < 8; ++j) {
    const int p = cq * 128 + j * 16 + (lane & 15);
#pragma unroll
    for (int r = 0; r < 8; ++r)
      attH[(size_t)(gRow + r) * P_DIM + p] = (f16)(acc[j][r] * cs[r]);
  }
}

// ---------------------------------------------------------------------------
// Kernel 5: output projection [16384,512] x [512,512] + bo, fp32 out.
// ---------------------------------------------------------------------------
__global__ void out_proj(const f16* __restrict__ attH, const f16* __restrict__ WoT,
                         const float* __restrict__ bo, float* __restrict__ out) {
  const int lane = threadIdx.x & 31;
  const int wave = threadIdx.x >> 5;
  const int rowBase = blockIdx.y * 32 + (wave & 1) * 16;
  const int colBase = blockIdx.x * 256 + (wave >> 1) * 64;

  v8f acc[4] = {v8f{}, v8f{}, v8f{}, v8f{}};
  const f16* aRow = attH + (size_t)(rowBase + (lane & 15)) * P_DIM;

  for (int pc = 0; pc < P_DIM; pc += 32) {
    v16h a = load_a_frag(aRow, pc, lane);
#pragma unroll
    for (int j = 0; j < 4; ++j) {
      const f16* bRow = WoT + (size_t)(colBase + j * 16 + (lane & 15)) * P_DIM;
      v16h b = load_bt_frag(bRow, pc, lane);
      acc[j] = wmma_f16(a, b, acc[j]);
    }
  }

  const int mBase = rowBase + ((lane >> 4) << 3);
#pragma unroll
  for (int j = 0; j < 4; ++j) {
    const int col = colBase + j * 16 + (lane & 15);
    const float bias = bo[col];
#pragma unroll
    for (int r = 0; r < 8; ++r)
      out[(size_t)(mBase + r) * D_DIM + col] = acc[j][r] + bias;
  }
}

// ---------------------------------------------------------------------------
extern "C" void kernel_launch(void* const* d_in, const int* in_sizes, int n_in,
                              void* d_out, int out_size, void* d_ws, size_t ws_size,
                              hipStream_t stream) {
  (void)in_sizes; (void)n_in; (void)out_size; (void)ws_size;
  const float* x  = (const float*)d_in[0];
  const float* Wq = (const float*)d_in[1];
  const float* bq = (const float*)d_in[2];
  const float* Wk = (const float*)d_in[3];
  const float* bk = (const float*)d_in[4];
  const float* Wv = (const float*)d_in[5];
  const float* bv = (const float*)d_in[6];
  const float* Wo = (const float*)d_in[7];
  const float* bo = (const float*)d_in[8];
  float* out = (float*)d_out;

  char* ws = (char*)d_ws;
  size_t off = 0;
  auto wsAlloc = [&](size_t bytes) -> void* {
    void* p = ws + off;
    off += (bytes + 255) & ~(size_t)255;
    return p;
  };

  f16* xh    = (f16*)wsAlloc((size_t)ROWS_TOTAL * D_DIM * 2);  // 16 MB
  f16* WqkvT = (f16*)wsAlloc((size_t)QKV_COLS * D_DIM * 2);    // 1.5 MB
  f16* WoT   = (f16*)wsAlloc((size_t)D_DIM * P_DIM * 2);       // 0.5 MB
  f16* qh    = (f16*)wsAlloc((size_t)ROWS_TOTAL * P_DIM * 2);  // 16 MB
  f16* kh    = (f16*)wsAlloc((size_t)ROWS_TOTAL * P_DIM * 2);  // 16 MB
  f16* vT    = (f16*)wsAlloc((size_t)ROWS_TOTAL * P_DIM * 2);  // 16 MB  [b][p][t]
  f16* attH  = (f16*)wsAlloc((size_t)ROWS_TOTAL * P_DIM * 2);  // 16 MB

  prep_x<<<(ROWS_TOTAL * D_DIM) / (256 * 4), 256, 0, stream>>>(x, xh);
  prep_weights<<<(D_DIM * P_DIM) / 256, 256, 0, stream>>>(Wq, Wk, Wv, Wo, WqkvT, WoT);
  qkv_gemm<<<dim3(QKV_COLS / 256, ROWS_TOTAL / 32), 256, 0, stream>>>(
      xh, WqkvT, bq, bk, bv, qh, kh, vT);
  attention<<<dim3(S_LEN / 32, BATCH), 256, 0, stream>>>(qh, kh, vT, attH);
  out_proj<<<dim3(D_DIM / 256, ROWS_TOTAL / 32), 256, 0, stream>>>(attH, WoT, bo, out);
}
